// QLSTMGen121_65481071395509
// MI455X (gfx1250) — compile-verified, matrix operand order checked by reading, not run on hardware
//
#include <hip/hip_runtime.h>
#include <hip/hip_bf16.h>

// ---------------------------------------------------------------------------
// QLSTM on MI455X (gfx1250, wave32, WMMA bf16 16x16x32, f32 accumulate)
//
// Phase 0: prep kernels pack x and W into WMMA operand layouts (bf16).
// Phase 1: parallel WMMA GEMM  Xpre[t] = x_t @ Wx^T + bias   (bf16 out, swizzled
//          into the 16x16 f32 C-matrix VGPR layout so phase 2 loads are v8 per lane).
// Phase 2: single persistent workgroup (16 waves). Recurrent weights Wh live in
//          VGPRs (v16bf wh[4][8] per wave = 256 VGPRs). h is exchanged via a
//          double-buffered LDS array each step; 64 v_wmma per wave per step.
// ---------------------------------------------------------------------------

typedef __attribute__((ext_vector_type(16))) __bf16 bf16x16;
typedef __attribute__((ext_vector_type(8)))  __bf16 bf16x8;
typedef __attribute__((ext_vector_type(8)))  float  f32x8;

#define T_TOT 2048
#define B_N   32
#define D_N   256
#define H_N   256
#define HSTR  264   // padded LDS row stride in halves (528B) to spread banks

// 16-bit WMMA operand packing rule (ISA 7.12.2, 16-bit A 16x32 table):
//   lane l: r = l%16 (row for A / column for B), half = l/16
//   element e in [0,16): k = (e<8) ? half*8+e : 16+half*8+(e-8)
__device__ __forceinline__ int pack_k(int e, int half) {
    return (e < 8) ? (half * 8 + e) : (16 + half * 8 + (e - 8));
}

// ---------------------------------------------------------------------------
// Prep: pack weights into B-operand tiles.
//   wx[ntile(64)][k(8)][lane(32)][16 halves]   (K = x part, cols 0..255 of W)
//   wh[ntile(64)][k(8)][lane(32)][16 halves]   (K = h part, cols 256..511)
// ntile = gate*16 + (hcol/16), gates: 0=f 1=i 2=g 3=o
// ---------------------------------------------------------------------------
__global__ void prep_w_kernel(const float* __restrict__ Wf, const float* __restrict__ Wi,
                              const float* __restrict__ Wg, const float* __restrict__ Wo,
                              __bf16* __restrict__ wx, __bf16* __restrict__ wh) {
    int idx = blockIdx.x * blockDim.x + threadIdx.x;   // 2*64*8*32 = 32768
    if (idx >= 2 * 64 * 8 * 32) return;
    int lane  = idx & 31;
    int k     = (idx >> 5) & 7;
    int ntile = (idx >> 8) & 63;
    int part  = idx >> 14;            // 0 = x part, 1 = h part
    int g  = ntile >> 4;
    int nt = ntile & 15;
    int r = lane & 15, half = lane >> 4;
    const float* W = (g == 0) ? Wf : (g == 1) ? Wi : (g == 2) ? Wg : Wo;
    const float* row = W + (size_t)(nt * 16 + r) * (D_N + H_N) + (part ? D_N : 0) + k * 32;
    union { __bf16 a[16]; bf16x16 v; } u;
#pragma unroll
    for (int e = 0; e < 16; e++) u.a[e] = (__bf16)row[pack_k(e, half)];
    __bf16* base = part ? wh : wx;
    *(bf16x16*)(base + (((size_t)ntile * 8 + k) * 32 + lane) * 16) = u.v;
}

// ---------------------------------------------------------------------------
// Prep: pack x[t0..t0+chunk) into A-operand tiles (bf16).
//   abuf[tl][mt(2)][k(8)][lane(32)][16 halves]
// ---------------------------------------------------------------------------
__global__ void prep_a_kernel(const float* __restrict__ x, __bf16* __restrict__ abuf,
                              int t0, int chunk) {
    int idx = blockIdx.x * blockDim.x + threadIdx.x;   // chunk*512
    if (idx >= chunk * 512) return;
    int lane = idx & 31;
    int k    = (idx >> 5) & 7;
    int mt   = (idx >> 8) & 1;
    int tl   = idx >> 9;
    int r = lane & 15, half = lane >> 4;
    int b = mt * 16 + r;
    const float* row = x + (((size_t)(t0 + tl) * B_N + b) * D_N) + k * 32;
    union { __bf16 a[16]; bf16x16 v; } u;
#pragma unroll
    for (int e = 0; e < 16; e++) u.a[e] = (__bf16)row[pack_k(e, half)];
    *(bf16x16*)(abuf + ((((size_t)tl * 2 + mt) * 8 + k) * 32 + lane) * 16) = u.v;
}

// ---------------------------------------------------------------------------
// Phase 1: Xpre = x @ Wx^T + bias, output bf16 in C-matrix lane layout:
//   xpre[tl][ntile(64)][mt(2)][lane(32)][8 halves]
// One wave per (tl, mt, ng); wave computes ntiles {g*16+ng, g=0..3}.
// ---------------------------------------------------------------------------
__global__ void lstm_xproj_kernel(const __bf16* __restrict__ abuf,
                                  const __bf16* __restrict__ wx,
                                  const float* __restrict__ bfv, const float* __restrict__ biv,
                                  const float* __restrict__ bgv, const float* __restrict__ bov,
                                  __bf16* __restrict__ xpre, int chunk) {
    int wid  = (blockIdx.x * blockDim.x + threadIdx.x) >> 5;  // global wave id
    int lane = threadIdx.x & 31;
    int ng = wid & 15;
    int mt = (wid >> 4) & 1;
    int tl = wid >> 5;
    if (tl >= chunk) return;  // wave-uniform guard

    f32x8 acc[4] = {};
#pragma unroll
    for (int k = 0; k < 8; k++) {
        bf16x16 a = *(const bf16x16*)(abuf + ((((size_t)tl * 2 + mt) * 8 + k) * 32 + lane) * 16);
#pragma unroll
        for (int g = 0; g < 4; g++) {
            int ntile = g * 16 + ng;
            bf16x16 b = *(const bf16x16*)(wx + (((size_t)ntile * 8 + k) * 32 + lane) * 16);
            acc[g] = __builtin_amdgcn_wmma_f32_16x16x32_bf16(
                false, a, false, b, (short)0, acc[g], false, false);
        }
    }
    int hc = ng * 16 + (lane & 15);
    const float* biases[4] = {bfv, biv, bgv, bov};
#pragma unroll
    for (int g = 0; g < 4; g++) {
        float bb = biases[g][hc];
        union { __bf16 a[8]; bf16x8 v; } u;
#pragma unroll
        for (int j = 0; j < 8; j++) u.a[j] = (__bf16)(acc[g][j] + bb);
        *(bf16x8*)(xpre + ((((size_t)tl * 64 + (g * 16 + ng)) * 2 + mt) * 32 + lane) * 8) = u.v;
    }
}

// ---------------------------------------------------------------------------
// Phase 2: sequential recurrence, single workgroup, 16 waves.
// Wave w owns hidden columns [16w, 16w+16) for all 4 gates (ntiles g*16+w).
// ---------------------------------------------------------------------------
__device__ __forceinline__ float sigm_f(float x) { return 1.0f / (1.0f + __expf(-x)); }
__device__ __forceinline__ float tanh_f(float x) {
    x = fminf(fmaxf(x, -15.0f), 15.0f);
    float e = __expf(2.0f * x);
    return (e - 1.0f) / (e + 1.0f);
}

__global__ __launch_bounds__(512) void lstm_rec_kernel(
    const __bf16* __restrict__ xpre, const __bf16* __restrict__ whp,
    float* __restrict__ out, __bf16* __restrict__ hstate, float* __restrict__ cstate,
    int t0, int chunk) {
    __shared__ __bf16 hbuf[2][B_N * HSTR];

    int lane = threadIdx.x & 31;
    int w    = threadIdx.x >> 5;          // wave id 0..15
    int rr   = lane & 15;
    int half = lane >> 4;
    int hc0  = w * 16 + rr;               // this lane's gate/hidden column

    // Recurrent weights resident in VGPRs for the whole kernel: 256 VGPRs/wave.
    bf16x16 wh[4][8];
#pragma unroll
    for (int g = 0; g < 4; g++)
#pragma unroll
        for (int k = 0; k < 8; k++)
            wh[g][k] = *(const bf16x16*)(whp + (((size_t)(g * 16 + w) * 8 + k) * 32 + lane) * 16);

    // Load carried h into LDS buffer 0 (padded layout).
    for (int i = threadIdx.x; i < B_N * H_N; i += 512) {
        int b = i >> 8, k = i & 255;
        hbuf[0][b * HSTR + k] = hstate[i];
    }
    // Load carried c slice into registers (C-matrix layout).
    f32x8 c[2];
#pragma unroll
    for (int mt = 0; mt < 2; mt++)
#pragma unroll
        for (int j = 0; j < 8; j++)
            c[mt][j] = cstate[(size_t)(mt * 16 + j + 8 * half) * H_N + hc0];
    __syncthreads();

    int cur = 0;
    for (int tl = 0; tl < chunk; ++tl) {
        int t = t0 + tl;
        // Pull next step's Xpre toward the caches early.
        if (tl + 1 < chunk)
            __builtin_prefetch(xpre + (((size_t)(tl + 1) * 64 + w) * 2) * 32 * 8, 0, 1);

#pragma unroll
        for (int mt = 0; mt < 2; mt++) {
            // Precomputed input projection for this step (issued early, f32-add later).
            bf16x8 xp[4];
#pragma unroll
            for (int g = 0; g < 4; g++)
                xp[g] = *(const bf16x8*)(xpre +
                         ((((size_t)tl * 64 + (g * 16 + w)) * 2 + mt) * 32 + lane) * 8);

            f32x8 acc[4] = {};
#pragma unroll
            for (int k = 0; k < 8; k++) {
                // A-tile of h from LDS: two ds_load_b128 per lane.
                int b = mt * 16 + rr;
                const __bf16* p = &hbuf[cur][b * HSTR + k * 32 + half * 8];
                union { uint4 q[2]; bf16x16 v; } ua;
                ua.q[0] = *(const uint4*)(p);
                ua.q[1] = *(const uint4*)(p + 16);
#pragma unroll
                for (int g = 0; g < 4; g++)
                    acc[g] = __builtin_amdgcn_wmma_f32_16x16x32_bf16(
                        false, ua.v, false, wh[g][k], (short)0, acc[g], false, false);
            }

            // Gate nonlinearity + state update (TRANS ops co-execute with XDL).
#pragma unroll
            for (int j = 0; j < 8; j++) {
                float gf = sigm_f(acc[0][j] + (float)xp[0][j]);
                float gi = sigm_f(acc[1][j] + (float)xp[1][j]);
                float gg = tanh_f(acc[2][j] + (float)xp[2][j]);
                float go = sigm_f(acc[3][j] + (float)xp[3][j]);
                float cn = gf * c[mt][j] + gi * gg;
                c[mt][j] = cn;
                float hn = go * tanh_f(cn);
                int b = mt * 16 + j + 8 * half;
                out[((size_t)t * B_N + b) * H_N + hc0] = hn;                  // stacked
                hbuf[cur ^ 1][b * HSTR + hc0] = (__bf16)hn;                   // next-step h
                if (t == T_TOT - 1)
                    out[(size_t)T_TOT * B_N * H_N + (size_t)b * H_N + hc0] = hn;  // hx
            }
        }
        __syncthreads();   // hbuf[cur^1] fully written, hbuf[cur] fully read
        cur ^= 1;
    }

    // Persist carried state for the next chunk.
    for (int i = threadIdx.x; i < B_N * H_N; i += 512) {
        int b = i >> 8, k = i & 255;
        hstate[i] = hbuf[cur][b * HSTR + k];
    }
#pragma unroll
    for (int mt = 0; mt < 2; mt++)
#pragma unroll
        for (int j = 0; j < 8; j++) {
            int b = mt * 16 + j + 8 * half;
            cstate[(size_t)b * H_N + hc0] = c[mt][j];
            if (t0 + chunk == T_TOT)
                out[(size_t)T_TOT * B_N * H_N + (size_t)B_N * H_N + (size_t)b * H_N + hc0] =
                    c[mt][j];   // cx
        }
}

// ---------------------------------------------------------------------------
// Host launcher
// ---------------------------------------------------------------------------
extern "C" void kernel_launch(void* const* d_in, const int* in_sizes, int n_in,
                              void* d_out, int out_size, void* d_ws, size_t ws_size,
                              hipStream_t stream) {
    const float* x   = (const float*)d_in[0];
    const float* Wf  = (const float*)d_in[1];
    const float* bfv = (const float*)d_in[2];
    const float* Wi  = (const float*)d_in[3];
    const float* biv = (const float*)d_in[4];
    const float* Wg  = (const float*)d_in[5];
    const float* bgv = (const float*)d_in[6];
    const float* Wo  = (const float*)d_in[7];
    const float* bov = (const float*)d_in[8];
    float* out = (float*)d_out;

    char* ws = (char*)d_ws;
    size_t off = 0;
    __bf16* wx  = (__bf16*)(ws + off); off += (size_t)64 * 8 * 32 * 32;   // 512 KB
    __bf16* whp = (__bf16*)(ws + off); off += (size_t)64 * 8 * 32 * 32;   // 512 KB
    size_t state_off = off;
    __bf16* hstate = (__bf16*)(ws + off); off += (size_t)B_N * H_N * 2;   // 16 KB
    float*  cstate = (float*) (ws + off); off += (size_t)B_N * H_N * 4;   // 32 KB
    off = (off + 255) & ~(size_t)255;

    // Choose chunk so abuf + xpre fit in remaining workspace.
    const size_t per_t = 16384 /*abuf*/ + 65536 /*xpre*/;
    size_t avail = (ws_size > off) ? (ws_size - off) : 0;
    int chunk = T_TOT;
    while (chunk > 1 && (size_t)chunk * per_t > avail) chunk >>= 1;

    __bf16* abuf = (__bf16*)(ws + off);
    __bf16* xpre = (__bf16*)(ws + off + (size_t)chunk * 16384);

    // Zero-init carried state (h0 = c0 = 0) every call (deterministic).
    hipMemsetAsync(ws + state_off, 0, (size_t)B_N * H_N * 2 + (size_t)B_N * H_N * 4, stream);

    prep_w_kernel<<<(2 * 64 * 8 * 32 + 255) / 256, 256, 0, stream>>>(Wf, Wi, Wg, Wo, wx, whp);

    for (int t0 = 0; t0 < T_TOT; t0 += chunk) {
        prep_a_kernel<<<(chunk * 512 + 255) / 256, 256, 0, stream>>>(x, abuf, t0, chunk);
        // chunk*32 waves total, 8 waves (256 threads) per block -> chunk*4 blocks
        lstm_xproj_kernel<<<chunk * 4, 256, 0, stream>>>(abuf, wx, bfv, biv, bgv, bov,
                                                         xpre, chunk);
        lstm_rec_kernel<<<1, 512, 0, stream>>>(xpre, whp, out, hstate, cstate, t0, chunk);
    }
}
